// GPT_61237643707005
// MI455X (gfx1250) — matrix operand order repaired and checked by reading, not compiled, not run on hardware
//
#include <hip/hip_runtime.h>
#include <hip/hip_bf16.h>
#include <math.h>

// =====================================================================
// GPT-2 small forward for MI455X (gfx1250), compile-only target.
//
// All GEMMs run through v_wmma_f32_16x16x32_bf16 (bf16 in, f32 accum).
// Tile staging uses GLOBAL_LOAD_ASYNC_TO_LDS_B128 (ASYNCcnt). The GEMM
// is templated on its flag set so each of the 6 model GEMM shapes gets
// a straight-line staging loop + specialized epilogue. Logits epilogue
// uses non-temporal stores (824 MB write-once stream; keep L2 clean).
// Attention = scores GEMM (causal-block-skip) -> causal softmax ->
// P@V GEMM (causal K clip). LayerNorm / softmax / residuals in f32.
// =====================================================================

namespace {
constexpr int NB   = 4;
constexpr int S    = 1024;
constexpr int D    = 768;
constexpr int H    = 12;
constexpr int DH   = 64;   // D / H
constexpr int NL   = 12;
constexpr int DFF  = 3072;
constexpr int D3   = 3 * D;
constexpr int ROWS = NB * S;   // 4096
}

typedef __bf16 bf16;
typedef __attribute__((ext_vector_type(16))) __bf16 v16bf;
typedef __attribute__((ext_vector_type(8)))  float  v8f;
typedef int vi4 __attribute__((vector_size(4 * sizeof(int))));

#define AS1 __attribute__((address_space(1)))
#define AS3 __attribute__((address_space(3)))

#if defined(__HIP_DEVICE_COMPILE__) && __has_builtin(__builtin_amdgcn_global_load_async_to_lds_b128)
#define HAVE_ASYNC_LDS 1
#else
#define HAVE_ASYNC_LDS 0
#endif

#if HAVE_ASYNC_LDS
__device__ __forceinline__ void async_copy_b128(const bf16* g, bf16* l) {
  // builtin signature (probe-confirmed):
  //   (v4i addrspace(1)*, v4i addrspace(3)*, imm offset, imm cpol)
  __builtin_amdgcn_global_load_async_to_lds_b128(
      (AS1 vi4*)(g), (AS3 vi4*)(l), /*offset=*/0, /*cpol=*/0);
}
__device__ __forceinline__ void wait_async() {
#if __has_builtin(__builtin_amdgcn_s_wait_asynccnt)
  __builtin_amdgcn_s_wait_asynccnt(0);
#else
  asm volatile("s_wait_asynccnt 0" ::: "memory");
#endif
}
#endif

enum : int {
  GF_BIAS   = 1,   // add f32 bias[col]
  GF_GELU   = 2,   // exact erf GELU epilogue
  GF_BF16   = 4,   // store C as bf16 (else f32)
  GF_TRANSB = 8,   // B given as [N x K] row-major (i.e. use B^T)
  GF_CSKIP  = 16,  // causal: skip C blocks entirely above the diagonal
  GF_CK     = 32,  // causal: clip K loop at (block_row_end) (for P@V)
  GF_NT     = 64,  // non-temporal C stores (write-once streams, e.g. logits)
};

#define BM 128
#define BN 128
#define BK 32
#define LSTR 40   // LDS row stride in bf16 (80 B: 16B aligned, bank-friendly)

// --------------------------------------------------------------------
// f32 -> bf16 bulk convert (weights, embeddings)
// --------------------------------------------------------------------
__global__ __launch_bounds__(256) void cvt_bf16_kernel(
    const float* __restrict__ in, bf16* __restrict__ out, long long n)
{
  long long i  = (long long)blockIdx.x * blockDim.x + threadIdx.x;
  long long st = (long long)gridDim.x * blockDim.x;
  for (; i < n; i += st) out[i] = (bf16)in[i];
}

// --------------------------------------------------------------------
// x = tok_emb[tokens] + pos_emb  (writes f32 residual + bf16 GEMM input)
// --------------------------------------------------------------------
__global__ __launch_bounds__(256) void embed_kernel(
    const int* __restrict__ tokens, const float* __restrict__ te,
    const float* __restrict__ pe, float* __restrict__ xf,
    bf16* __restrict__ xb)
{
  const int row = blockIdx.x;            // 0..ROWS-1 (b*S + s)
  const int tok = tokens[row];
  const int pos = row % S;
  const long long base = (long long)row * D;
  for (int c = threadIdx.x; c < D; c += blockDim.x) {
    float v = te[(long long)tok * D + c] + pe[(long long)pos * D + c];
    xf[base + c] = v;
    xb[base + c] = (bf16)v;
  }
}

// --------------------------------------------------------------------
// block reductions (256 threads = 8 waves of 32)
// --------------------------------------------------------------------
__device__ __forceinline__ float block_sum(float v, float* sm) {
  __syncthreads();
  #pragma unroll
  for (int o = 16; o > 0; o >>= 1) v += __shfl_xor(v, o, 32);
  if ((threadIdx.x & 31) == 0) sm[threadIdx.x >> 5] = v;
  __syncthreads();
  if (threadIdx.x < 32) {
    float t = (threadIdx.x < 8) ? sm[threadIdx.x] : 0.0f;
    #pragma unroll
    for (int o = 4; o > 0; o >>= 1) t += __shfl_xor(t, o, 32);
    if (threadIdx.x == 0) sm[0] = t;
  }
  __syncthreads();
  return sm[0];
}

__device__ __forceinline__ float block_max(float v, float* sm) {
  __syncthreads();
  #pragma unroll
  for (int o = 16; o > 0; o >>= 1) v = fmaxf(v, __shfl_xor(v, o, 32));
  if ((threadIdx.x & 31) == 0) sm[threadIdx.x >> 5] = v;
  __syncthreads();
  if (threadIdx.x < 32) {
    float t = (threadIdx.x < 8) ? sm[threadIdx.x] : -3.0e38f;
    #pragma unroll
    for (int o = 4; o > 0; o >>= 1) t = fmaxf(t, __shfl_xor(t, o, 32));
    if (threadIdx.x == 0) sm[0] = t;
  }
  __syncthreads();
  return sm[0];
}

// --------------------------------------------------------------------
// out = LayerNorm(a + b) * w + bias ; dual f32/bf16 output (either optional)
// one block per row, D = 768 (3 elems/thread)
// --------------------------------------------------------------------
__global__ __launch_bounds__(256) void ln_residual_kernel(
    const float* __restrict__ a, const float* __restrict__ b,
    const float* __restrict__ w, const float* __restrict__ bias,
    float* __restrict__ outF, bf16* __restrict__ outB)
{
  __shared__ float sm[8];
  const long long base = (long long)blockIdx.x * D;
  float v[3]; int cs[3];
  float s = 0.0f;
  #pragma unroll
  for (int i = 0; i < 3; ++i) {
    int c = threadIdx.x + i * 256; cs[i] = c;
    float t = 0.0f;
    if (c < D) { t = a[base + c]; if (b) t += b[base + c]; }
    v[i] = t; s += t;
  }
  const float mu = block_sum(s, sm) * (1.0f / D);
  float q = 0.0f;
  #pragma unroll
  for (int i = 0; i < 3; ++i)
    if (cs[i] < D) { float d = v[i] - mu; q += d * d; }
  const float rstd = rsqrtf(block_sum(q, sm) * (1.0f / D) + 1e-5f);
  #pragma unroll
  for (int i = 0; i < 3; ++i) {
    int c = cs[i];
    if (c < D) {
      float y = (v[i] - mu) * rstd * w[c] + bias[c];
      if (outF) outF[base + c] = y;
      if (outB) outB[base + c] = (bf16)y;
    }
  }
}

// --------------------------------------------------------------------
// causal softmax over one row of [S] f32 scores -> bf16 probs
// grid.x = NB*H*S rows; masked cols (c > s) are never read, written as 0
// --------------------------------------------------------------------
__global__ __launch_bounds__(256) void softmax_causal_kernel(
    const float* __restrict__ scores, bf16* __restrict__ probs)
{
  __shared__ float sm[8];
  const long long row = blockIdx.x;
  const int sidx = (int)(row % S);
  const float* in = scores + row * (long long)S;
  bf16* out = probs + row * (long long)S;

  float x[4];                       // S = 1024 = 4 * 256
  float mx = -3.0e38f;
  #pragma unroll
  for (int i = 0; i < 4; ++i) {
    int c = threadIdx.x + i * 256;
    x[i] = (c <= sidx) ? in[c] : -3.0e38f;
    mx = fmaxf(mx, x[i]);
  }
  mx = block_max(mx, sm);
  float e[4], acc = 0.0f;
  #pragma unroll
  for (int i = 0; i < 4; ++i) {
    int c = threadIdx.x + i * 256;
    e[i] = (c <= sidx) ? __expf(x[i] - mx) : 0.0f;
    acc += e[i];
  }
  acc = block_sum(acc, sm);
  const float inv = 1.0f / acc;
  #pragma unroll
  for (int i = 0; i < 4; ++i) {
    int c = threadIdx.x + i * 256;
    out[c] = (bf16)(e[i] * inv);
  }
}

// --------------------------------------------------------------------
// Generic batched bf16 WMMA GEMM, templated on FLAGS:
//   C[z][M,N] = scale * (A[z][M,K] @ B[z][K,N]) (+bias) (GELU) -> f32|bf16
// z = blockIdx.z, decomposed as z2 = z/zdiv, z1 = z%zdiv with independent
// strides (lets us address per-(batch,head) Q/K/V slices of the qkv buffer).
//
// REQUIREMENTS (all satisfied by this model's shapes):
//   M % 128 == 0 (4096, 1024); K % 32 == 0; for !TRANSB: N % 8 == 0.
//
// Block = 256 threads = 8 waves; tile BMxBN = 128x128, BK = 32.
// Wave grid 4x2; each wave owns 32x64 = 2x4 v_wmma_f32_16x16x32_bf16 accums.
//
// Fragment layouts per CDNA5 ISA 7.12.2 (wave32):
//   A 16x32 bf16: lane m = lane%16; elems 0..7  = K (lane<16?0:8)+0..7,
//                                   elems 8..15 = 16 + (lane<16?0:8)+0..7
//   B 32x16 bf16: lane n = lane%16; same K order as A
//   C 16x16 f32 : VGPR r -> row r + (lane<16?0:8), col lane%16
// LDS stores A as [m][k] and B as [n][k], row stride 40 bf16 (16B aligned),
// so every fragment load is 1-2 contiguous b128 LDS reads.
// --------------------------------------------------------------------
template <int FLAGS>
__global__ __launch_bounds__(256) void gemm_bf16_wmma_kernel(
    const bf16* __restrict__ A, const bf16* __restrict__ B,
    const float* __restrict__ bias, void* __restrict__ Cout,
    int M, int N, int K, int lda, int ldb, int ldc,
    float scale, int zdiv,
    long long sAz1, long long sAz2,
    long long sBz1, long long sBz2,
    long long sCz1, long long sCz2)
{
  __shared__ __align__(16) bf16 Asm[BM * LSTR];
  __shared__ __align__(16) bf16 Bsm[BN * LSTR];

  const int z  = blockIdx.z;
  const int z1 = z % zdiv;
  const int z2 = z / zdiv;
  A += (long long)z1 * sAz1 + (long long)z2 * sAz2;
  B += (long long)z1 * sBz1 + (long long)z2 * sBz2;
  const long long cbase = (long long)z1 * sCz1 + (long long)z2 * sCz2;

  const int bm0 = blockIdx.y * BM;
  const int bn0 = blockIdx.x * BN;
  if ((FLAGS & GF_CSKIP) && bn0 > bm0 + (BM - 1)) return;  // fully masked

  int kEnd = K;
  if (FLAGS & GF_CK) { int km = bm0 + BM; kEnd = km < K ? km : K; }
  kEnd = (kEnd + BK - 1) / BK * BK;   // K is a multiple of 32 in all uses

  const int tid  = threadIdx.x;
  const int wave = tid >> 5;
  const int lane = tid & 31;
  const int hh   = lane >> 4;   // which 16-lane half
  const int l16  = lane & 15;
  const int wm   = wave >> 1;   // 0..3
  const int wn   = wave & 1;    // 0..1

  const bool fullBN = (bn0 + BN <= N);   // uniform across the block

  v8f acc[2][4];
  #pragma unroll
  for (int i = 0; i < 2; ++i)
    #pragma unroll
    for (int j = 0; j < 4; ++j) acc[i][j] = {};

  for (int k0 = 0; k0 < kEnd; k0 += BK) {
    // ---- stage A tile (BM x BK): 512 b128 chunks / 256 threads ----
    // M % 128 == 0 -> no row guard.
    #pragma unroll
    for (int i = 0; i < 2; ++i) {
      int c  = tid + i * 256;
      int r  = c >> 2;
      int cc = (c & 3) * 8;
      const bf16* ap = A + (long long)(bm0 + r) * lda + k0 + cc;
#if HAVE_ASYNC_LDS
      async_copy_b128(ap, &Asm[r * LSTR + cc]);
#else
      *(uint4*)(&Asm[r * LSTR + cc]) = *(const uint4*)ap;
      if (k0 + BK < kEnd) __builtin_prefetch(ap + BK, 0, 1);
#endif
    }
    // ---- stage B tile as [n][k] ----
    if (FLAGS & GF_TRANSB) {
      // B is [N x K] row-major: already k-contiguous per n
      #pragma unroll
      for (int i = 0; i < 2; ++i) {
        int c  = tid + i * 256;
        int n  = c >> 2;
        int kk = (c & 3) * 8;
        int gn = bn0 + n;
        const bf16* bp = B + (long long)gn * ldb + k0 + kk;
#if HAVE_ASYNC_LDS
        if (fullBN) {
          async_copy_b128(bp, &Bsm[n * LSTR + kk]);
          continue;
        }
#endif
        uint4 v = make_uint4(0u, 0u, 0u, 0u);
        if (gn < N) v = *(const uint4*)bp;
        *(uint4*)(&Bsm[n * LSTR + kk]) = v;
      }
    } else {
      // B is [K x N] row-major: read n-contiguous, scatter transposed.
      // N % 8 == 0 for all !TRANSB uses -> an 8-wide chunk is either
      // fully in-bounds or fully out-of-bounds (no per-element tail).
      #pragma unroll
      for (int i = 0; i < 2; ++i) {
        int c  = tid + i * 256;
        int kr = c >> 4;
        int nc = (c & 15) * 8;
        int gn = bn0 + nc;
        union { uint4 q; bf16 e[8]; } v;
        const bf16* bp = B + (long long)(k0 + kr) * ldb + gn;
        if (fullBN) {                    // uniform: straight-line path
          v.q = *(const uint4*)bp;
        } else {
          v.q = make_uint4(0u, 0u, 0u, 0u);
          if (gn + 8 <= N) v.q = *(const uint4*)bp;
        }
        #pragma unroll
        for (int j = 0; j < 8; ++j) Bsm[(nc + j) * LSTR + kr] = v.e[j];
      }
    }
#if HAVE_ASYNC_LDS
    wait_async();          // drain ASYNCcnt before the workgroup barrier
#endif
    __syncthreads();

    // ---- fragments + WMMA ----
    union Frag { v16bf v; uint4 q[2]; } af[2], bfr[4];
    #pragma unroll
    for (int tm = 0; tm < 2; ++tm) {
      const bf16* p = &Asm[(wm * 32 + tm * 16 + l16) * LSTR + hh * 8];
      af[tm].q[0] = *(const uint4*)(p);
      af[tm].q[1] = *(const uint4*)(p + 16);
    }
    #pragma unroll
    for (int tn = 0; tn < 4; ++tn) {
      const bf16* p = &Bsm[(wn * 64 + tn * 16 + l16) * LSTR + hh * 8];
      bfr[tn].q[0] = *(const uint4*)(p);
      bfr[tn].q[1] = *(const uint4*)(p + 16);
    }
    #pragma unroll
    for (int tm = 0; tm < 2; ++tm)
      #pragma unroll
      for (int tn = 0; tn < 4; ++tn)
        acc[tm][tn] = __builtin_amdgcn_wmma_f32_16x16x32_bf16(
            false, af[tm].v, false, bfr[tn].v,
            (short)0, acc[tm][tn], false, false);
    __syncthreads();
  }

  // ---- epilogue ----
  float* Cf = (float*)Cout;
  bf16*  Ch = (bf16*)Cout;
  #pragma unroll
  for (int tm = 0; tm < 2; ++tm) {
    #pragma unroll
    for (int tn = 0; tn < 4; ++tn) {
      const int col = bn0 + wn * 64 + tn * 16 + l16;
      if (col >= N) continue;
      const float bv = (FLAGS & GF_BIAS) ? bias[col] : 0.0f;
      #pragma unroll
      for (int r = 0; r < 8; ++r) {
        const int row = bm0 + wm * 32 + tm * 16 + hh * 8 + r;
        float v = acc[tm][tn][r] * scale + bv;
        if (FLAGS & GF_GELU) v = 0.5f * v * (1.0f + erff(v * 0.70710678f));
        const long long idx = cbase + (long long)row * ldc + col;
        if (FLAGS & GF_BF16) {
          Ch[idx] = (bf16)v;
        } else if (FLAGS & GF_NT) {
          __builtin_nontemporal_store(v, Cf + idx);  // write-once stream
        } else {
          Cf[idx] = v;
        }
      }
    }
  }
}

// --------------------------------------------------------------------
// host-side launch helpers (compile-time flag dispatch)
// --------------------------------------------------------------------
template <int F>
static void launch_gemm_t(hipStream_t st,
    const bf16* A, const bf16* B, const float* bias, void* C,
    int M, int N, int K, int lda, int ldb, int ldc, float scale,
    int Z, int zdiv,
    long long sA1, long long sA2, long long sB1, long long sB2,
    long long sC1, long long sC2)
{
  dim3 g((N + BN - 1) / BN, (M + BM - 1) / BM, Z);
  gemm_bf16_wmma_kernel<F><<<g, 256, 0, st>>>(A, B, bias, C, M, N, K,
      lda, ldb, ldc, scale, zdiv, sA1, sA2, sB1, sB2, sC1, sC2);
}

static void launch_gemm(hipStream_t st,
    const bf16* A, const bf16* B, const float* bias, void* C,
    int M, int N, int K, int lda, int ldb, int ldc,
    float scale, int flags,
    int Z = 1, int zdiv = 1,
    long long sA1 = 0, long long sA2 = 0,
    long long sB1 = 0, long long sB2 = 0,
    long long sC1 = 0, long long sC2 = 0)
{
  switch (flags) {
#define GEMM_CASE(F) case (F): \
    launch_gemm_t<(F)>(st, A, B, bias, C, M, N, K, lda, ldb, ldc, scale, \
                       Z, zdiv, sA1, sA2, sB1, sB2, sC1, sC2); break;
    GEMM_CASE(GF_BIAS | GF_BF16)                 // qkv projection
    GEMM_CASE(GF_TRANSB | GF_CSKIP)              // attention scores
    GEMM_CASE(GF_BF16 | GF_CK)                   // P @ V
    GEMM_CASE(GF_BIAS)                           // Wo, FF2
    GEMM_CASE(GF_BIAS | GF_GELU | GF_BF16)       // FF1 + GELU
    GEMM_CASE(GF_TRANSB | GF_NT)                 // logits (tied embedding)
#undef GEMM_CASE
    default: break;  // unused combination
  }
}

extern "C" void kernel_launch(void* const* d_in, const int* in_sizes, int n_in,
                              void* d_out, int out_size, void* d_ws, size_t ws_size,
                              hipStream_t stream)
{
  const int*   tokens  = (const int*)  d_in[0];
  const float* tok_emb = (const float*)d_in[1];
  const float* pos_emb = (const float*)d_in[2];
  const float* Wqkv    = (const float*)d_in[3];
  const float* bqkv    = (const float*)d_in[4];
  const float* Wo      = (const float*)d_in[5];
  const float* bo      = (const float*)d_in[6];
  const float* ln1w    = (const float*)d_in[7];
  const float* ln1b    = (const float*)d_in[8];
  const float* W1      = (const float*)d_in[9];
  const float* b1      = (const float*)d_in[10];
  const float* W2      = (const float*)d_in[11];
  const float* b2      = (const float*)d_in[12];
  const float* ln2w    = (const float*)d_in[13];
  const float* ln2b    = (const float*)d_in[14];
  const int NV = in_sizes[1] / D;   // 50257

  // ---- workspace carve-up (~620 MB; MI455X has 432 GiB) ----
  char* p = (char*)d_ws;
  auto alloc = [&](size_t bytes) -> void* {
    void* r = (void*)p; p += (bytes + 255) & ~(size_t)255; return r;
  };
  bf16*  wqkv_b  = (bf16*) alloc((size_t)NL * D * D3 * 2);
  bf16*  wo_b    = (bf16*) alloc((size_t)NL * D * D  * 2);
  bf16*  w1_b    = (bf16*) alloc((size_t)NL * D * DFF * 2);
  bf16*  w2_b    = (bf16*) alloc((size_t)NL * DFF * D * 2);
  bf16*  emb_b   = (bf16*) alloc((size_t)NV * D * 2);
  float* x_f     = (float*)alloc((size_t)ROWS * D * 4);
  bf16*  x_b     = (bf16*) alloc((size_t)ROWS * D * 2);
  bf16*  qkv_b   = (bf16*) alloc((size_t)ROWS * D3 * 2);
  float* scor_f  = (float*)alloc((size_t)NB * H * S * S * 4);
  bf16*  prob_b  = (bf16*) alloc((size_t)NB * H * S * S * 2);
  bf16*  att_b   = (bf16*) alloc((size_t)ROWS * D * 2);
  float* mha_f   = (float*)alloc((size_t)ROWS * D * 4);
  bf16*  h_b     = (bf16*) alloc((size_t)ROWS * D * 2);
  bf16*  ffm_b   = (bf16*) alloc((size_t)ROWS * DFF * 2);
  float* ff_f    = (float*)alloc((size_t)ROWS * D * 4);

  // ---- one-time f32 -> bf16 conversion of all matmul operands ----
  cvt_bf16_kernel<<<4096, 256, 0, stream>>>(Wqkv,    wqkv_b, (long long)NL * D * D3);
  cvt_bf16_kernel<<<4096, 256, 0, stream>>>(Wo,      wo_b,   (long long)NL * D * D);
  cvt_bf16_kernel<<<4096, 256, 0, stream>>>(W1,      w1_b,   (long long)NL * D * DFF);
  cvt_bf16_kernel<<<4096, 256, 0, stream>>>(W2,      w2_b,   (long long)NL * DFF * D);
  cvt_bf16_kernel<<<4096, 256, 0, stream>>>(tok_emb, emb_b,  (long long)NV * D);

  // ---- embeddings ----
  embed_kernel<<<ROWS, 256, 0, stream>>>(tokens, tok_emb, pos_emb, x_f, x_b);

  const long long SS = (long long)S * S;
  for (int l = 0; l < NL; ++l) {
    const bf16* wq = wqkv_b + (size_t)l * D * D3;
    const bf16* wo = wo_b   + (size_t)l * D * D;
    const bf16* w1 = w1_b   + (size_t)l * D * DFF;
    const bf16* w2 = w2_b   + (size_t)l * DFF * D;

    // 1) qkv = x @ Wqkv + bqkv           [4096 x 2304], bf16 out
    launch_gemm(stream, x_b, wq, bqkv + (size_t)l * D3, qkv_b,
                ROWS, D3, D, D, D3, D3, 1.0f, GF_BIAS | GF_BF16);

    // 2) scores = Q @ K^T / 8            per (b,h): [1024 x 1024] f32
    //    z = b*H + h ; Q col offset h*64, K col offset 768 + h*64
    launch_gemm(stream, qkv_b, qkv_b + D, nullptr, scor_f,
                S, S, DH, D3, D3, S, 0.125f, GF_TRANSB | GF_CSKIP,
                NB * H, H,
                /*A*/ DH, (long long)S * D3,
                /*B*/ DH, (long long)S * D3,
                /*C*/ SS, (long long)H * SS);

    // 3) causal softmax -> bf16 probs
    softmax_causal_kernel<<<NB * H * S, 256, 0, stream>>>(scor_f, prob_b);

    // 4) att = P @ V                     per (b,h): [1024 x 64], bf16 out
    launch_gemm(stream, prob_b, qkv_b + 2 * D, nullptr, att_b,
                S, DH, S, S, D3, D, 1.0f, GF_BF16 | GF_CK,
                NB * H, H,
                /*A*/ SS, (long long)H * SS,
                /*B*/ DH, (long long)S * D3,
                /*C*/ DH, (long long)S * D);

    // 5) mha = att @ Wo + bo             [4096 x 768] f32
    launch_gemm(stream, att_b, wo, bo + (size_t)l * D, mha_f,
                ROWS, D, D, D, D, D, 1.0f, GF_BIAS);

    // 6) h = LN(mha + x)                 bf16 only (FF input)
    ln_residual_kernel<<<ROWS, 256, 0, stream>>>(
        mha_f, x_f, ln1w + (size_t)l * D, ln1b + (size_t)l * D, nullptr, h_b);

    // 7) ffmid = GELU(h @ W1 + b1)       [4096 x 3072] bf16
    launch_gemm(stream, h_b, w1, b1 + (size_t)l * DFF, ffm_b,
                ROWS, DFF, D, D, DFF, DFF, 1.0f, GF_BIAS | GF_GELU | GF_BF16);

    // 8) ff = ffmid @ W2 + b2            [4096 x 768] f32
    launch_gemm(stream, ffm_b, w2, b2 + (size_t)l * D, ff_f,
                ROWS, D, DFF, DFF, D, D, 1.0f, GF_BIAS);

    // 9) x = LN(ff + x)   (second residual adds LAYER INPUT x, per reference)
    ln_residual_kernel<<<ROWS, 256, 0, stream>>>(
        ff_f, x_f, ln2w + (size_t)l * D, ln2b + (size_t)l * D, x_f, x_b);
  }

  // logits = x @ tok_emb^T               [4096 x 50257] f32 -> d_out
  launch_gemm(stream, x_b, emb_b, nullptr, d_out,
              ROWS, NV, D, D, D, NV, 1.0f, GF_TRANSB | GF_NT);
}
